// Euc_branch_36395552866672
// MI455X (gfx1250) — compile-verified, matrix operand order checked by reading, not run on hardware
//
#include <hip/hip_runtime.h>

// ---------------------------------------------------------------------------
// MI455X (gfx1250) 3-D U-Net forward.
// Bandwidth-bound (~100 GFLOP vs ~5 GB of activation traffic -> HBM-limited),
// so: GEMMs on the WMMA pipe (f16 in / f32 accum), weights L2-resident,
// tap-major K loop (no per-element division), float4 streaming epilogues,
// TDM (tensor_load_to_lds) staging for the rectangular convT A-tiles.
// ---------------------------------------------------------------------------

typedef __attribute__((ext_vector_type(16))) _Float16 v16h;
typedef __attribute__((ext_vector_type(8)))  _Float16 v8h;
typedef __attribute__((ext_vector_type(8)))  float    v8f;
typedef __attribute__((ext_vector_type(4))) unsigned int u32x4;
typedef __attribute__((ext_vector_type(8))) int          i32x8;
typedef __attribute__((ext_vector_type(4))) int          i32x4;

#define LDS_STRIDE 40   // f16 row stride: 80B, keeps each 8-halfword group 16B aligned

// ---------------------------------------------------------------------------
// Implicit-GEMM conv: out[m,n] = sum_{tap,ci} A[m, tap*Ci+ci] * W[tap*Ci+ci, n]
// Tap-major K loop: no integer division in the hot path; per-row voxel coords
// precomputed once into LDS. 128 threads = 4 waves; block owns a 32-row M
// tile; each wave owns a 16-col N tile and issues two WMMAs per chunk
// (rows 0-15 / 16-31) sharing one B fragment.
// Handles ks=3/pad=1 (SAME), ks=2/stride=2 (down), ks=1 (GEMM: lin/logits).
// ---------------------------------------------------------------------------
__global__ __launch_bounds__(128) void conv_wmma_kernel(
    const float* __restrict__ in, const float* __restrict__ w,
    const float* __restrict__ bias, float* __restrict__ out,
    int Din, int Hin, int Win, int Ci,
    int Dout, int Hout, int Wout, int Co,
    int ks, int stride, int pad)
{
  const int tid  = threadIdx.x;
  const int lane = tid & 31;
  const int wave = tid >> 5;
  const int Mtot  = Dout * Hout * Wout;
  const int Mbase = blockIdx.x << 5;                // 32 rows per block
  const int Nbase = (blockIdx.y * 4 + wave) << 4;
  const bool active = (Nbase < Co);                 // wave-uniform
  const int Ktot = ks * ks * ks * Ci;

  __shared__ alignas(16) _Float16 As[32 * LDS_STRIDE];
  __shared__ int rz[32], ry[32], rx[32];

  // per-row voxel coordinates (one division per row, once per block)
  if (tid < 32) {
    int m = Mbase + tid;
    if (m < Mtot) {
      int mz = m / (Hout * Wout); int m2 = m - mz * Hout * Wout;
      int my = m2 / Wout;         int mx = m2 - my * Wout;
      rz[tid] = mz * stride - pad;
      ry[tid] = my * stride - pad;
      rx[tid] = mx * stride - pad;
    } else {
      rz[tid] = -1000000; ry[tid] = 0; rx[tid] = 0;  // forces zero-fill
    }
  }
  __syncthreads();

  v8f acc0 = {}, acc1 = {};
  const int n    = Nbase + (lane & 15);
  const int koff = (lane >> 4) << 3;     // CDNA5 16-bit A/B lane layout
  const int row  = lane & 15;
  const bool nval = (n < Co);

  for (int dz = 0; dz < ks; ++dz)
  for (int dy = 0; dy < ks; ++dy)
  for (int dx = 0; dx < ks; ++dx) {
    const int wtap = ((dz * ks + dy) * ks + dx) * Ci;
    for (int cc = 0; cc < Ci; cc += 32) {
      // ---- stage A tile (32 rows x 32 ci) into LDS as f16 ----
#pragma unroll
      for (int j = 0; j < 8; ++j) {
        int e  = tid + (j << 7);
        int r  = e >> 5;
        int kl = e & 31;
        int ci = cc + kl;
        int iz = rz[r] + dz, iy = ry[r] + dy, ix = rx[r] + dx;
        float v = 0.0f;
        if (ci < Ci && (unsigned)iz < (unsigned)Din &&
            (unsigned)iy < (unsigned)Hin && (unsigned)ix < (unsigned)Win)
          v = in[((iz * Hin + iy) * Win + ix) * Ci + ci];
        As[r * LDS_STRIDE + kl] = (_Float16)v;
      }
      __syncthreads();

      if (active) {
        const _Float16* rp = &As[row * LDS_STRIDE];
        v8h lo0 = *reinterpret_cast<const v8h*>(rp + koff);
        v8h hi0 = *reinterpret_cast<const v8h*>(rp + 16 + koff);
        v16h a0 = __builtin_shufflevector(lo0, hi0,
                    0,1,2,3,4,5,6,7,8,9,10,11,12,13,14,15);
        rp += 16 * LDS_STRIDE;
        v8h lo1 = *reinterpret_cast<const v8h*>(rp + koff);
        v8h hi1 = *reinterpret_cast<const v8h*>(rp + 16 + koff);
        v16h a1 = __builtin_shufflevector(lo1, hi1,
                    0,1,2,3,4,5,6,7,8,9,10,11,12,13,14,15);

        v16h b;   // weights: hot in the 192MB L2
#pragma unroll
        for (int jj = 0; jj < 8; ++jj) {
          int ci0 = cc + koff + jj;
          int ci1 = ci0 + 16;
          float b0 = (ci0 < Ci && nval) ? w[(wtap + ci0) * Co + n] : 0.0f;
          float b1 = (ci1 < Ci && nval) ? w[(wtap + ci1) * Co + n] : 0.0f;
          b[jj]     = (_Float16)b0;
          b[jj + 8] = (_Float16)b1;
        }
        if (nval && (wtap + cc + 32 + koff) < Ktot)
          __builtin_prefetch(&w[(wtap + cc + 32 + koff) * Co + n], 0, 1);

        acc0 = __builtin_amdgcn_wmma_f32_16x16x32_f16(
                 false, a0, false, b, (short)0, acc0, false, false);
        acc1 = __builtin_amdgcn_wmma_f32_16x16x32_f16(
                 false, a1, false, b, (short)0, acc1, false, false);
      }
      __syncthreads();
    }
  }

  if (active && nval) {
    float bb = bias ? bias[n] : 0.0f;
    int mb = Mbase + ((lane >> 4) << 3);
#pragma unroll
    for (int r = 0; r < 8; ++r) {
      int m0 = mb + r;
      if (m0 < Mtot) out[m0 * Co + n] = acc0[r] + bb;
      int m1 = m0 + 16;
      if (m1 < Mtot) out[m1 * Co + n] = acc1[r] + bb;
    }
  }
}

// ---------------------------------------------------------------------------
// Transposed conv 2x2x2 stride-2 == 8 plain GEMMs (one per output parity):
//   out[2z+pd,2y+ph,2x+pw,:] = in[z,y,x,:] @ W[pd,ph,pw,:,:]
// The A operand is a rectangular [32 x 32] f32 tile with row stride Ci ->
// staged by the Tensor Data Mover (tensor_load_to_lds + s_wait_tensorcnt),
// with tensor_dim clamping giving free zero-fill at the M/K edges.
// ---------------------------------------------------------------------------
__global__ __launch_bounds__(128) void convt_wmma_kernel(
    const float* __restrict__ in, const float* __restrict__ w,
    float* __restrict__ out, int Gin, int Ci, int Co)
{
  const int tid  = threadIdx.x;
  const int lane = tid & 31;
  const int wave = tid >> 5;
  const int Min   = Gin * Gin * Gin;
  const int Mbase = blockIdx.x << 5;
  const int Nbase = (blockIdx.y * 4 + wave) << 4;
  const bool active = (Nbase < Co);
  const int p  = blockIdx.z;
  const int pd = p >> 2, ph = (p >> 1) & 1, pw = p & 1;
  const int Gout  = Gin << 1;
  const int wbase = p * Ci;

  __shared__ alignas(16) float Af[32 * 32];

  v8f acc0 = {}, acc1 = {};
  const int n    = Nbase + (lane & 15);
  const int koff = (lane >> 4) << 3;
  const int row  = lane & 15;
  const bool nval = (n < Co);

  for (int kk = 0; kk < Ci; kk += 32) {
#if defined(__gfx1250__) && __has_builtin(__builtin_amdgcn_tensor_load_to_lds)
    if (tid < 32) {   // one wave drives the TDM; descriptor is all-uniform
      unsigned long long ga =
          (unsigned long long)(const void*)(in + Mbase * Ci + kk);
      unsigned lds = (unsigned)(unsigned long long)(const void*)Af;
      int td0 = Ci - kk;     // remaining K columns  -> OOB reads return 0
      int td1 = Min - Mbase; // remaining M rows     -> OOB reads return 0
      u32x4 g0;
      g0[0] = 1u;                                   // count=1, user mode
      g0[1] = lds;                                  // lds_addr
      g0[2] = (unsigned)ga;                         // global_addr[31:0]
      g0[3] = (unsigned)((ga >> 32) & 0x01FFFFFFu) | 0x80000000u; // addr|type=2
      i32x8 g1;
      g1[0] = 0x20000;                              // data_size=4B, no mcast
      g1[1] = (td0 & 0xFFFF) << 16;                 // tensor_dim0 lo
      g1[2] = ((td0 >> 16) & 0xFFFF) | ((td1 & 0xFFFF) << 16);
      g1[3] = ((td1 >> 16) & 0xFFFF) | (32 << 16);  // tile_dim0 = 32
      g1[4] = 32;                                   // tile_dim1 = 32
      g1[5] = Ci;                                   // tensor_dim0_stride lo
      g1[6] = 0;
      g1[7] = 0;
      i32x4 z4 = {0, 0, 0, 0};
#if __clang_major__ >= 23
      i32x8 z8 = {0, 0, 0, 0, 0, 0, 0, 0};
      __builtin_amdgcn_tensor_load_to_lds(g0, g1, z4, z4, z8, 0);
#else
      __builtin_amdgcn_tensor_load_to_lds(g0, g1, z4, z4, 0);
#endif
      __builtin_amdgcn_s_wait_tensorcnt(0);
    }
#else
    // fallback: cooperative staging
#pragma unroll
    for (int j = 0; j < 8; ++j) {
      int e = tid + (j << 7);
      int r = e >> 5, kl = e & 31;
      int k = kk + kl, m = Mbase + r;
      Af[r * 32 + kl] = (k < Ci && m < Min) ? in[m * Ci + k] : 0.0f;
    }
#endif
    __syncthreads();

    if (active) {
      const float* rp0 = &Af[row * 32 + koff];
      const float* rp1 = rp0 + 16 * 32;
      v16h a0, a1;
#pragma unroll
      for (int j = 0; j < 8; ++j) {
        a0[j]     = (_Float16)rp0[j];
        a0[j + 8] = (_Float16)rp0[16 + j];
        a1[j]     = (_Float16)rp1[j];
        a1[j + 8] = (_Float16)rp1[16 + j];
      }
      v16h b;
#pragma unroll
      for (int jj = 0; jj < 8; ++jj) {
        int k0 = kk + koff + jj;
        int k1 = k0 + 16;
        float b0 = (k0 < Ci && nval) ? w[(wbase + k0) * Co + n] : 0.0f;
        float b1 = (k1 < Ci && nval) ? w[(wbase + k1) * Co + n] : 0.0f;
        b[jj]     = (_Float16)b0;
        b[jj + 8] = (_Float16)b1;
      }
      acc0 = __builtin_amdgcn_wmma_f32_16x16x32_f16(
               false, a0, false, b, (short)0, acc0, false, false);
      acc1 = __builtin_amdgcn_wmma_f32_16x16x32_f16(
               false, a1, false, b, (short)0, acc1, false, false);
    }
    __syncthreads();
  }

  if (active && nval) {
    int mb = Mbase + ((lane >> 4) << 3);
#pragma unroll
    for (int r = 0; r < 8; ++r) {
#pragma unroll
      for (int t = 0; t < 2; ++t) {
        int m = mb + r + t * 16;
        if (m < Min) {
          int z = m / (Gin * Gin); int m2 = m - z * Gin * Gin;
          int y = m2 / Gin;        int x  = m2 - y * Gin;
          int oi = ((z * 2 + pd) * Gout + (y * 2 + ph)) * Gout + (x * 2 + pw);
          out[oi * Co + n] = (t ? acc1[r] : acc0[r]);
        }
      }
    }
  }
}

// ---------------------------------------------------------------------------
// BN stats, deterministic two-stage reduction (bit-stable across replays).
// Stage 1: block = (channel-quad, slice); float4 reads; LDS tree reduce.
// Stage 2: per-channel serial combine of slices + fused scale/shift.
// ---------------------------------------------------------------------------
__global__ __launch_bounds__(256) void bn_stats_part_kernel(
    const float* __restrict__ x, int nvox, int C, int nslice,
    float* __restrict__ part)
{
  const int c4 = blockIdx.x << 2;
  float s0=0,s1=0,s2=0,s3=0,q0=0,q1=0,q2=0,q3=0;
  for (int v = blockIdx.y * 256 + threadIdx.x; v < nvox; v += nslice * 256) {
    const float4 t = *reinterpret_cast<const float4*>(x + v * C + c4);
    s0 += t.x; q0 += t.x * t.x;
    s1 += t.y; q1 += t.y * t.y;
    s2 += t.z; q2 += t.z * t.z;
    s3 += t.w; q3 += t.w * t.w;
  }
  __shared__ float sh[8 * 256];
  sh[0*256+threadIdx.x]=s0; sh[1*256+threadIdx.x]=s1;
  sh[2*256+threadIdx.x]=s2; sh[3*256+threadIdx.x]=s3;
  sh[4*256+threadIdx.x]=q0; sh[5*256+threadIdx.x]=q1;
  sh[6*256+threadIdx.x]=q2; sh[7*256+threadIdx.x]=q3;
  __syncthreads();
  for (int o = 128; o > 0; o >>= 1) {
    if (threadIdx.x < o) {
#pragma unroll
      for (int k = 0; k < 8; ++k)
        sh[k*256+threadIdx.x] += sh[k*256+threadIdx.x+o];
    }
    __syncthreads();
  }
  if (threadIdx.x < 8)
    part[(blockIdx.x * nslice + blockIdx.y) * 8 + threadIdx.x] =
        sh[threadIdx.x * 256];
}

__global__ void bn_stats_fin_kernel(
    const float* __restrict__ part, int nslice, int nvox, int C,
    const float* __restrict__ g, const float* __restrict__ b,
    float* __restrict__ ss)
{
  int c = blockIdx.x * blockDim.x + threadIdx.x;
  if (c >= C) return;
  int cq = c >> 2, l = c & 3;
  float s = 0.0f, q = 0.0f;
  for (int j = 0; j < nslice; ++j) {
    s += part[(cq * nslice + j) * 8 + l];
    q += part[(cq * nslice + j) * 8 + 4 + l];
  }
  float invN = 1.0f / (float)nvox;
  float mean = s * invN;
  float var  = q * invN - mean * mean;
  float sc   = g[c] * rsqrtf(var + 1e-5f);
  ss[c]     = sc;
  ss[C + c] = b[c] - mean * sc;
}

// relu(x*scale+shift) (+ residual), float4 streams, channel-offset store
// (lets the up-conv write directly into the concat buffer).
__global__ void bn_apply_kernel(
    const float* __restrict__ x, const float* __restrict__ ss,
    const float* __restrict__ res, float* __restrict__ out,
    int nvox, int C, int outC, int outOff)
{
  int total = (nvox * C) >> 2;
  for (int i = blockIdx.x * blockDim.x + threadIdx.x; i < total;
       i += gridDim.x * blockDim.x) {
    int e = i << 2;
    int v = e / C, c = e - v * C;
    float4 t  = *reinterpret_cast<const float4*>(x + e);
    float4 sc = *reinterpret_cast<const float4*>(ss + c);
    float4 sf = *reinterpret_cast<const float4*>(ss + C + c);
    float4 r;
    r.x = fmaxf(t.x * sc.x + sf.x, 0.0f);
    r.y = fmaxf(t.y * sc.y + sf.y, 0.0f);
    r.z = fmaxf(t.z * sc.z + sf.z, 0.0f);
    r.w = fmaxf(t.w * sc.w + sf.w, 0.0f);
    if (res) {
      float4 rr = *reinterpret_cast<const float4*>(res + e);
      r.x += rr.x; r.y += rr.y; r.z += rr.z; r.w += rr.w;
    }
    *reinterpret_cast<float4*>(out + v * outC + outOff + c) = r;
  }
}

__global__ void copy_ch_kernel(
    const float* __restrict__ src, float* __restrict__ dst,
    int nvox, int C, int dstC, int dstOff)
{
  int total = (nvox * C) >> 2;
  for (int i = blockIdx.x * blockDim.x + threadIdx.x; i < total;
       i += gridDim.x * blockDim.x) {
    int e = i << 2;
    int v = e / C, c = e - v * C;
    *reinterpret_cast<float4*>(dst + v * dstC + dstOff + c) =
        *reinterpret_cast<const float4*>(src + e);
  }
}

// ---------------------------------------------------------------------------
// Host orchestration
// ---------------------------------------------------------------------------
static inline int cdiv(int a, int b) { return (a + b - 1) / b; }
static inline int hmin(int a, int b) { return a < b ? a : b; }
static inline int hmax(int a, int b) { return a > b ? a : b; }

static void launch_conv(hipStream_t s, const float* in, const float* w,
                        const float* bias, float* out,
                        int Gin, int Ci, int Gout, int Co,
                        int ks, int stride, int pad) {
  dim3 grid(cdiv(Gout * Gout * Gout, 32), cdiv(cdiv(Co, 16), 4), 1);
  conv_wmma_kernel<<<grid, 128, 0, s>>>(in, w, bias, out,
                                        Gin, Gin, Gin, Ci,
                                        Gout, Gout, Gout, Co,
                                        ks, stride, pad);
}

static void launch_bn(hipStream_t s, const float* x, int nvox, int C,
                      const float* g, const float* b, float* ss, float* part,
                      const float* res, float* out, int outC, int outOff) {
  int nslice = hmin(32, hmax(1, nvox / 256));
  dim3 gs(C >> 2, nslice);
  bn_stats_part_kernel<<<gs, 256, 0, s>>>(x, nvox, C, nslice, part);
  bn_stats_fin_kernel<<<cdiv(C, 64), 64, 0, s>>>(part, nslice, nvox, C, g, b, ss);
  int total = (nvox * C) >> 2;
  bn_apply_kernel<<<hmin(cdiv(total, 256), 4096), 256, 0, s>>>(
      x, ss, res, out, nvox, C, outC, outOff);
}

extern "C" void kernel_launch(void* const* d_in, const int* in_sizes, int n_in,
                              void* d_out_v, int out_size, void* d_ws, size_t ws_size,
                              hipStream_t stream) {
  (void)in_sizes; (void)n_in; (void)out_size; (void)ws_size;

  const float* features = (const float*)d_in[0];   // (1,64,64,64,3) dense grid
  // d_in[1] = indices: dense row-major grid -> unused
  int pi = 2;
  auto P = [&]() { return (const float*)d_in[pi++]; };

  const float* input_w = P();
  const float* input_g = P();
  const float* input_b = P();

  const float *en_w1[6], *en_g1[6], *en_b1[6], *en_w2[6], *en_g2[6], *en_b2[6];
  const float *dn_w[6], *dn_g[6], *dn_b[6];
  for (int i = 0; i < 6; ++i) {
    en_w1[i] = P(); en_g1[i] = P(); en_b1[i] = P();
    en_w2[i] = P(); en_g2[i] = P(); en_b2[i] = P();
    dn_w[i]  = P(); dn_g[i]  = P(); dn_b[i]  = P();
  }
  const float* mid_w1 = P(); const float* mid_g1 = P(); const float* mid_b1 = P();
  const float* mid_w2 = P(); const float* mid_g2 = P(); const float* mid_b2 = P();

  const float *up_w[6], *up_g[6], *up_b[6], *lin_w[6];
  const float *de_w1[6], *de_g1[6], *de_b1[6], *de_w2[6], *de_g2[6], *de_b2[6];
  for (int i = 0; i < 6; ++i) {
    up_w[i] = P(); up_g[i] = P(); up_b[i] = P(); lin_w[i] = P();
    de_w1[i] = P(); de_g1[i] = P(); de_b1[i] = P();
    de_w2[i] = P(); de_g2[i] = P(); de_b2[i] = P();
  }
  const float* out_w = P();
  const float* out_b = P();

  float* d_out = (float*)d_out_v;
  char*  ws    = (char*)d_ws;
  size_t off   = 0;
  auto alloc = [&](size_t nfloats) -> float* {
    float* p = (float*)(ws + off);
    off += ((nfloats * 4 + 255) & ~(size_t)255);
    return p;
  };

  const int CHS[7] = {32, 64, 96, 128, 160, 192, 224};
  int Gd[7], V[7];
  for (int i = 0; i < 7; ++i) { Gd[i] = 64 >> i; V[i] = Gd[i] * Gd[i] * Gd[i]; }

  float* skip[6];
  for (int i = 0; i < 6; ++i) skip[i] = alloc((size_t)V[i] * CHS[i]);
  float* bufH = alloc(8388608);     // 64^3 * 32
  float* bufT = alloc(8388608);
  float* bufR = alloc(8388608);
  float* bufC = alloc(16777216);    // 64^3 * 64 (largest concat)
  float* ss   = alloc(512);         // fused scale/shift (C <= 224)
  float* part = alloc(14336);       // (224/4)*32 slices * 8 partials

  // d_out layout: logits, x_mid, y5, y4, y3, y2, y1, y0
  const int off_xmid = V[0] * 20;
  int off_y[6];
  {
    int o = off_xmid + 224;
    for (int i = 5; i >= 0; --i) { off_y[i] = o; o += V[i] * CHS[i]; }
  }

  // ---------------- stem ----------------
  launch_conv(stream, features, input_w, nullptr, bufT, 64, 3, 64, 32, 3, 1, 1);
  launch_bn(stream, bufT, V[0], 32, input_g, input_b, ss, part, nullptr, bufH, 32, 0);

  // ---------------- encoder ----------------
  for (int i = 0; i < 6; ++i) {
    int c = CHS[i], G = Gd[i], Vi = V[i];
    launch_conv(stream, bufH, en_w1[i], nullptr, bufT, G, c, G, c, 3, 1, 1);
    launch_bn(stream, bufT, Vi, c, en_g1[i], en_b1[i], ss, part, nullptr, bufR, c, 0);
    launch_conv(stream, bufR, en_w2[i], nullptr, bufT, G, c, G, c, 3, 1, 1);
    launch_bn(stream, bufT, Vi, c, en_g2[i], en_b2[i], ss, part, bufH, skip[i], c, 0);
    int cn = CHS[i + 1], Gn = Gd[i + 1], Vn = V[i + 1];
    launch_conv(stream, skip[i], dn_w[i], nullptr, bufT, G, c, Gn, cn, 2, 2, 0);
    launch_bn(stream, bufT, Vn, cn, dn_g[i], dn_b[i], ss, part, nullptr, bufH, cn, 0);
  }

  // ---------------- mid ----------------
  launch_conv(stream, bufH, mid_w1, nullptr, bufT, 1, 224, 1, 224, 3, 1, 1);
  launch_bn(stream, bufT, 1, 224, mid_g1, mid_b1, ss, part, nullptr, bufR, 224, 0);
  launch_conv(stream, bufR, mid_w2, nullptr, bufT, 1, 224, 1, 224, 3, 1, 1);
  launch_bn(stream, bufT, 1, 224, mid_g2, mid_b2, ss, part, bufH, d_out + off_xmid, 224, 0);

  // ---------------- decoder ----------------
  for (int i = 5; i >= 0; --i) {
    int ci = CHS[i + 1], co = CHS[i];
    int Gin = Gd[i + 1], Gout = Gd[i], Vo = V[i];
    const float* yin = (i == 5) ? (d_out + off_xmid) : (d_out + off_y[i + 1]);

    // up-conv (8 parity GEMMs, TDM-staged A) -> BN+ReLU into concat[:, :co]
    dim3 gt(cdiv(Gin * Gin * Gin, 32), cdiv(cdiv(co, 16), 4), 8);
    convt_wmma_kernel<<<gt, 128, 0, stream>>>(yin, up_w[i], bufT, Gin, ci, co);
    launch_bn(stream, bufT, Vo, co, up_g[i], up_b[i], ss, part, nullptr, bufC, 2 * co, 0);

    // concat[:, co:2co] = skip
    int tot = (Vo * co) >> 2;
    copy_ch_kernel<<<hmin(cdiv(tot, 256), 4096), 256, 0, stream>>>(
        skip[i], bufC, Vo, co, 2 * co, co);

    // lin residual branch: cat @ lin_w (1x1x1 GEMM)
    launch_conv(stream, bufC, lin_w[i], nullptr, bufR, Gout, 2 * co, Gout, co, 1, 1, 0);

    // de conv1 + BN
    launch_conv(stream, bufC, de_w1[i], nullptr, bufT, Gout, 2 * co, Gout, co, 3, 1, 1);
    launch_bn(stream, bufT, Vo, co, de_g1[i], de_b1[i], ss, part, nullptr, bufH, co, 0);

    // de conv2 + BN + residual(lin) -> y_i directly in d_out
    launch_conv(stream, bufH, de_w2[i], nullptr, bufT, Gout, co, Gout, co, 3, 1, 1);
    launch_bn(stream, bufT, Vo, co, de_g2[i], de_b2[i], ss, part, bufR, d_out + off_y[i], co, 0);
  }

  // ---------------- logits: y0 @ out_w + out_b ----------------
  launch_conv(stream, d_out + off_y[0], out_w, out_b, d_out, 64, 32, 64, 20, 1, 1, 0);
}
